// Generator_32959579030357
// MI455X (gfx1250) — matrix operand order repaired and checked by reading, not compiled, not run on hardware
//
#include <hip/hip_runtime.h>
#include <math.h>

#define B_    2
#define T_    512
#define S_    400
#define H_    1024
#define V_    50257
#define BT_   (B_ * T_)
#define NBLK_ 197
#define VPAD_ (NBLK_ * 256)   // 50432

typedef __attribute__((ext_vector_type(16))) __bf16 v16bf;
typedef __attribute__((ext_vector_type(8)))  __bf16 v8bf;
typedef __attribute__((ext_vector_type(8)))  float  v8f;

__device__ __forceinline__ __bf16 f2bf(float f) {
  unsigned u = __float_as_uint(f);
  u += 0x7FFFu + ((u >> 16) & 1u);            // round-to-nearest-even
  unsigned short h = (unsigned short)(u >> 16);
  return __builtin_bit_cast(__bf16, h);
}

// ---------------- conversion kernels ----------------
__global__ void cvt_x_kernel(const float* __restrict__ x, __bf16* __restrict__ xb, int n) {
  int i = blockIdx.x * blockDim.x + threadIdx.x;
  int stride = gridDim.x * blockDim.x;
  for (; i < n; i += stride) xb[i] = f2bf(x[i]);
}

__global__ void cvt_w_kernel(const float* __restrict__ w, __bf16* __restrict__ wb) {
  const long long ndst = (long long)VPAD_ * H_;
  const long long nsrc = (long long)V_ * H_;
  long long i = blockIdx.x * (long long)blockDim.x + threadIdx.x;
  long long stride = (long long)gridDim.x * blockDim.x;
  for (; i < ndst; i += stride)
    wb[i] = (i < nsrc) ? f2bf(w[i]) : f2bf(0.0f);   // zero-pad rows >= V
}

// ---------------- p_gen = sigmoid(x . pgen_w + pgen_b), one wave32 per row ----------------
__global__ __launch_bounds__(128) void pgen_kernel(const float* __restrict__ x,
                                                   const float* __restrict__ pw,
                                                   const float* __restrict__ pb,
                                                   float* __restrict__ pg) {
  int wave = threadIdx.x >> 5;
  int lane = threadIdx.x & 31;
  int row  = blockIdx.x * 4 + wave;
  if (row >= BT_) return;
  const float* xr = x + (size_t)row * H_;
  float sum = 0.0f;
  for (int k = lane; k < H_; k += 32) sum += xr[k] * pw[k];
  for (int m = 16; m >= 1; m >>= 1) sum += __shfl_xor(sum, m, 32);
  if (lane == 0) pg[row] = 1.0f / (1.0f + __expf(-(sum + pb[0])));
}

// ---------------- fragment loaders (contiguous k order, same for A and B) ----------------
__device__ __forceinline__ v16bf load_afrag(const __bf16* p) {
  // elems 0..15 -> K = half*16+(0..15): contiguous
  v8bf lo = *(const v8bf*)(p);
  v8bf hi = *(const v8bf*)(p + 8);
  return __builtin_shufflevector(lo, hi,
      0, 1, 2, 3, 4, 5, 6, 7, 8, 9, 10, 11, 12, 13, 14, 15);
}
__device__ __forceinline__ v16bf load_bfrag(const __bf16* p) {
  // lane holds col N=(lane&15); elems 0..15 -> K = (lane>>4)*16+(0..15): contiguous
  return *(const v16bf*)(p);
}

#define WMMA_BF16(A, Bm, C) \
  __builtin_amdgcn_wmma_f32_16x16x32_bf16(false, (A), false, (Bm), (short)0, (C), false, false)

// ---------------- WMMA bf16 GEMM: logit[bt, v] = x . proj_w^T + proj_b ----------------
// Block = 128 threads (4 waves). All 4 waves share the same 32 rows (A hits WGP$);
// each wave owns a distinct 64-col strip: 2 A frags x 4 B frags -> 8 accumulators.
// K loop is software-pipelined (register double-buffering) so WMMAs overlap the
// next iteration's L2 loads instead of s_wait_loadcnt 0 before every WMMA.
__global__ __launch_bounds__(128) void gemm_kernel(const __bf16* __restrict__ xb,
                                                   const __bf16* __restrict__ wb,
                                                   const float* __restrict__ bias,
                                                   float* __restrict__ out) {
  const int lane = threadIdx.x & 31;
  const int wave = threadIdx.x >> 5;
  const int mlo  = lane & 15;
  const int half = lane >> 4;
  const int row_base = blockIdx.y * 32;
  const int n_base   = blockIdx.x * 256 + wave * 64;

  const __bf16* aptr0 = xb + (size_t)(row_base + mlo) * H_ + half * 16;
  const __bf16* aptr1 = aptr0 + 16 * H_;
  const __bf16* bptr  = wb + (size_t)(n_base + mlo) * H_ + half * 16;

  v8f acc[8] = {v8f{}, v8f{}, v8f{}, v8f{}, v8f{}, v8f{}, v8f{}, v8f{}};

  // prologue: k = 0 fragments
  v16bf a0 = load_afrag(aptr0);
  v16bf a1 = load_afrag(aptr1);
  v16bf b0 = load_bfrag(bptr);
  v16bf b1 = load_bfrag(bptr + 16 * H_);
  v16bf b2 = load_bfrag(bptr + 32 * H_);
  v16bf b3 = load_bfrag(bptr + 48 * H_);

  #pragma unroll 4
  for (int k0 = 0; k0 < H_ - 32; k0 += 32) {
    const int kn = k0 + 32;
    // issue next-iteration loads first (double buffer in registers)
    v16bf a0n = load_afrag(aptr0 + kn);
    v16bf a1n = load_afrag(aptr1 + kn);
    v16bf b0n = load_bfrag(bptr + kn);
    v16bf b1n = load_bfrag(bptr + 16 * H_ + kn);
    v16bf b2n = load_bfrag(bptr + 32 * H_ + kn);
    v16bf b3n = load_bfrag(bptr + 48 * H_ + kn);

    acc[0] = WMMA_BF16(a0, b0, acc[0]);
    acc[1] = WMMA_BF16(a0, b1, acc[1]);
    acc[2] = WMMA_BF16(a0, b2, acc[2]);
    acc[3] = WMMA_BF16(a0, b3, acc[3]);
    acc[4] = WMMA_BF16(a1, b0, acc[4]);
    acc[5] = WMMA_BF16(a1, b1, acc[5]);
    acc[6] = WMMA_BF16(a1, b2, acc[6]);
    acc[7] = WMMA_BF16(a1, b3, acc[7]);

    a0 = a0n; a1 = a1n;
    b0 = b0n; b1 = b1n; b2 = b2n; b3 = b3n;
  }
  // epilogue: last k-step
  acc[0] = WMMA_BF16(a0, b0, acc[0]);
  acc[1] = WMMA_BF16(a0, b1, acc[1]);
  acc[2] = WMMA_BF16(a0, b2, acc[2]);
  acc[3] = WMMA_BF16(a0, b3, acc[3]);
  acc[4] = WMMA_BF16(a1, b0, acc[4]);
  acc[5] = WMMA_BF16(a1, b1, acc[5]);
  acc[6] = WMMA_BF16(a1, b2, acc[6]);
  acc[7] = WMMA_BF16(a1, b3, acc[7]);

  // C/D layout: lane -> N = mlo, VGPR r -> M = half*8 + r
  #pragma unroll
  for (int mi = 0; mi < 2; ++mi) {
    #pragma unroll
    for (int t = 0; t < 4; ++t) {
      int n = n_base + t * 16 + mlo;
      if (n < V_) {
        float bsum = bias[n];
        size_t base = (size_t)(row_base + mi * 16 + half * 8) * V_ + (size_t)n;
        #pragma unroll
        for (int r = 0; r < 8; ++r)
          out[base + (size_t)r * V_] = acc[mi * 4 + t][r] + bsum;
      }
    }
  }
}

// ---------------- row softmax over V (online, single read pass) * p_gen, in place ----------------
__global__ __launch_bounds__(256) void softmax_kernel(float* __restrict__ out,
                                                      const float* __restrict__ pg,
                                                      const int* __restrict__ temp) {
  __shared__ float sm[256];
  __shared__ float ss[256];
  const int row = blockIdx.x;
  const int tid = threadIdx.x;
  const float invt = 1.0f / (float)temp[0];
  float* r = out + (size_t)row * V_;

  float m = -3.4e38f, s = 0.0f;
  for (int n = tid; n < V_; n += 256) {
    float v = r[n] * invt;
    if (v > m) { s = s * __expf(m - v) + 1.0f; m = v; }
    else       { s += __expf(v - m); }
  }
  sm[tid] = m; ss[tid] = s;
  __syncthreads();
  for (int w = 128; w >= 1; w >>= 1) {
    if (tid < w) {
      float m2 = sm[tid + w], s2 = ss[tid + w];
      float M  = fmaxf(sm[tid], m2);
      ss[tid] = ss[tid] * __expf(sm[tid] - M) + s2 * __expf(m2 - M);
      sm[tid] = M;
    }
    __syncthreads();
  }
  const float M = sm[0];
  const float scale = pg[row] / ss[0];
  for (int n = tid; n < V_; n += 256)
    r[n] = __expf(r[n] * invt - M) * scale;
}

// ---------------- attention softmax over S + scatter-add into the vocab row ----------------
__global__ __launch_bounds__(128) void scatter_kernel(const float* __restrict__ attn,
                                                      const int* __restrict__ enc,
                                                      const float* __restrict__ pg,
                                                      const int* __restrict__ temp,
                                                      float* __restrict__ out) {
  __shared__ float red[128];
  const int row = blockIdx.x;          // row = b*T + t
  const int b   = row / T_;
  const int tid = threadIdx.x;
  const float invt = 1.0f / (float)temp[0];
  const float* a  = attn + (size_t)row * S_;
  const int*   ix = enc  + (size_t)b * S_;

  float m = -3.4e38f;
  for (int s = tid; s < S_; s += 128) m = fmaxf(m, a[s] * invt);
  red[tid] = m; __syncthreads();
  for (int w = 64; w >= 1; w >>= 1) {
    if (tid < w) red[tid] = fmaxf(red[tid], red[tid + w]);
    __syncthreads();
  }
  const float M = red[0];
  __syncthreads();

  float sum = 0.0f;
  for (int s = tid; s < S_; s += 128) sum += __expf(a[s] * invt - M);
  red[tid] = sum; __syncthreads();
  for (int w = 64; w >= 1; w >>= 1) {
    if (tid < w) red[tid] += red[tid + w];
    __syncthreads();
  }
  const float coef = (1.0f - pg[row]) / red[0];

  float* orow = out + (size_t)row * V_;
  for (int s = tid; s < S_; s += 128) {
    float val = __expf(a[s] * invt - M) * coef;
    unsafeAtomicAdd(&orow[ix[s]], val);
  }
}

// ---------------- final log ----------------
__global__ void log_kernel(float* __restrict__ out, long long n) {
  long long i = blockIdx.x * (long long)blockDim.x + threadIdx.x;
  long long stride = (long long)gridDim.x * blockDim.x;
  for (; i < n; i += stride) out[i] = __logf(out[i] + 1e-40f);
}

extern "C" void kernel_launch(void* const* d_in, const int* in_sizes, int n_in,
                              void* d_out, int out_size, void* d_ws, size_t ws_size,
                              hipStream_t stream) {
  const float* x      = (const float*)d_in[0];
  const float* attn   = (const float*)d_in[1];
  const int*   enc    = (const int*)d_in[2];
  const int*   temp   = (const int*)d_in[3];
  const float* proj_w = (const float*)d_in[4];
  const float* proj_b = (const float*)d_in[5];
  const float* pgen_w = (const float*)d_in[6];
  const float* pgen_b = (const float*)d_in[7];
  float* out = (float*)d_out;

  // workspace layout: bf16 weights (padded) | bf16 x | p_gen
  char* ws = (char*)d_ws;
  __bf16* wb = (__bf16*)ws;
  __bf16* xb = (__bf16*)(ws + (size_t)VPAD_ * H_ * 2);
  float*  pg = (float*)(ws + (size_t)VPAD_ * H_ * 2 + (size_t)BT_ * H_ * 2);

  cvt_w_kernel<<<8192, 256, 0, stream>>>(proj_w, wb);
  cvt_x_kernel<<<2048, 256, 0, stream>>>(x, xb, BT_ * H_);
  pgen_kernel<<<BT_ / 4, 128, 0, stream>>>(x, pgen_w, pgen_b, pg);
  gemm_kernel<<<dim3(NBLK_, BT_ / 32), 128, 0, stream>>>(xb, wb, proj_b, out);
  softmax_kernel<<<BT_, 256, 0, stream>>>(out, pg, temp);
  scatter_kernel<<<BT_, 128, 0, stream>>>(attn, enc, pg, temp, out);
  log_kernel<<<8192, 256, 0, stream>>>(out, (long long)BT_ * V_);
}
